// VectorQuantizer_35983236006234
// MI455X (gfx1250) — compile-verified, matrix operand order checked by reading, not run on hardware
//
#include <hip/hip_runtime.h>
#include <hip/hip_bf16.h>
#include <math.h>

typedef __attribute__((ext_vector_type(16))) _Float16 v16h;
typedef __attribute__((ext_vector_type(8)))  float    v8f;

#define Bq 32
#define Dq 256
#define Lq 1024
#define Kq 1024
#define Nq (Bq * Lq)   // 32768 rows

union FragU { v16h h; uint4 q[2]; };

// ---------------- prep: codebook fp32 -> f16, half-norms, zero accumulators ----
__global__ void vq_prep(const float* __restrict__ cb,
                        _Float16* __restrict__ cb16,
                        float* __restrict__ hnorm,
                        float* __restrict__ counts,
                        float* __restrict__ sumsq) {
    int k = blockIdx.x * blockDim.x + threadIdx.x;
    if (k < Kq) {
        const float* row = cb + (size_t)k * Dq;
        _Float16* orow = cb16 + (size_t)k * Dq;
        float nrm = 0.f;
        for (int d = 0; d < Dq; ++d) {
            float v = row[d];
            nrm += v * v;
            orow[d] = (_Float16)v;
        }
        hnorm[k]  = 0.5f * nrm;
        counts[k] = 0.f;
        if (k == 0) *sumsq = 0.f;
    }
}

// ---------------- main: fused WMMA distance + argmin + gather + reductions ----
__global__ __launch_bounds__(256) void vq_main(const float* __restrict__ zin,
                                               const float* __restrict__ cb32,
                                               const _Float16* __restrict__ cb16,
                                               const float* __restrict__ hnorm,
                                               float* __restrict__ counts,
                                               float* __restrict__ sumsq,
                                               float* __restrict__ zq_out,
                                               float* __restrict__ idx_out) {
    __shared__ __align__(16) _Float16 zlds[8][16][Dq];  // 64 KB
    __shared__ int bk[8][16];

    const int t    = threadIdx.x;
    const int w    = t >> 5;        // wave id in block (8 waves)
    const int lane = t & 31;
    const int row  = lane & 15;     // matrix row (A) / column (B,C) per WMMA layout
    const int sel  = lane >> 4;     // which half of the K-chunk this lane holds
    const int n0   = (blockIdx.x * 8 + w) * 16;  // first flat row for this wave
    const int b    = n0 >> 10;      // batch index (16 rows never cross b: 1024 % 16 == 0)
    const int l0   = n0 & 1023;

    // ---- stage 16 rows x 256 d of z into LDS as f16 (coalesced along L) ----
    {
        const float* zb = zin + ((size_t)b * Dq) * Lq + (l0 + row);
        for (int i = 0; i < Dq / 2; ++i) {
            int d = sel + 2 * i;
            zlds[w][row][d] = (_Float16)zb[(size_t)d * Lq];
        }
    }
    __syncthreads();

    // ---- preload 8 A-fragments (covering D=256 in 8 chunks of 32) ----
    // 16-bit A 16x32 layout: lane<16 holds K 0-7 & 16-23; lane>=16 holds 8-15 & 24-31
    FragU afrag[8];
#pragma unroll
    for (int c = 0; c < 8; ++c) {
        const _Float16* ap = &zlds[w][row][(c << 5) + sel * 8];
        afrag[c].q[0] = *(const uint4*)(ap);
        afrag[c].q[1] = *(const uint4*)(ap + 16);
    }

    float best[8];
    int   bkr[8];
#pragma unroll
    for (int r = 0; r < 8; ++r) { best[r] = -3.0e38f; bkr[r] = 0; }

    const int col = row;  // B/C column within the 16-wide code tile
    for (int kt = 0; kt < Kq / 16; ++kt) {
        const int k0 = kt << 4;
        const _Float16* brow = cb16 + ((size_t)(k0 + col) << 8);
        if (kt + 1 < Kq / 16)
            __builtin_prefetch(cb16 + ((size_t)(k0 + 16 + col) << 8), 0, 1);

        v8f acc = {0.f, 0.f, 0.f, 0.f, 0.f, 0.f, 0.f, 0.f};
#pragma unroll
        for (int c = 0; c < 8; ++c) {
            FragU bf;
            const _Float16* bp = brow + (c << 5) + sel * 8;
            bf.q[0] = *(const uint4*)(bp);
            bf.q[1] = *(const uint4*)(bp + 16);
            acc = __builtin_amdgcn_wmma_f32_16x16x32_f16(
                /*neg_a=*/false, afrag[c].h, /*neg_b=*/false, bf.h,
                /*c_mod=*/(short)0, acc, /*reuse_a=*/false, /*reuse_b=*/false);
        }

        // score = dot(z,e) - 0.5*||e||^2 ; argmax(score) == argmin(dist)
        const float hn    = hnorm[k0 + col];
        const int   kcode = k0 + col;
#pragma unroll
        for (int r = 0; r < 8; ++r) {
            float s = acc[r] - hn;
            if (s > best[r]) { best[r] = s; bkr[r] = kcode; }
        }
    }

    // ---- cross-lane argmax within each 16-lane half (C layout: lane = column) ----
#pragma unroll
    for (int r = 0; r < 8; ++r) {
        float s = best[r];
        int   k = bkr[r];
#pragma unroll
        for (int off = 1; off < 16; off <<= 1) {
            float os = __shfl_xor(s, off, 32);
            int   ok = __shfl_xor(k, off, 32);
            if (os > s || (os == s && ok < k)) { s = os; k = ok; }  // ties -> lowest k
        }
        const int m = r + (sel << 3);          // row index 0..15 within the tile
        if (col == 0) {                        // lanes 0 and 16 write
            bk[w][m] = k;
            idx_out[n0 + m] = (float)k;
        }
    }
    __syncthreads();

    // ---- gather z_q (fp32 codebook), write transposed output, loss + histogram ----
    const int   myk  = bk[w][row];
    const float* crw = cb32 + (size_t)myk * Dq;
    float* obase = zq_out + ((size_t)b * Dq) * Lq + (l0 + row);
    float ss = 0.f;
    for (int i = 0; i < Dq / 2; ++i) {
        int d = sel + 2 * i;
        float c  = crw[d];
        float z  = (float)zlds[w][row][d];
        float df = c - z;
        ss += df * df;
        obase[(size_t)d * Lq] = c;
    }
    if (sel == 0) atomicAdd(&counts[myk], 1.0f);  // one thread per row
#pragma unroll
    for (int off = 16; off >= 1; off >>= 1) ss += __shfl_xor(ss, off, 32);
    if (lane == 0) atomicAdd(sumsq, ss);
}

// ---------------- finalize: perplexity + vq_loss scalars ----------------------
__global__ void vq_fin(const float* __restrict__ counts,
                       const float* __restrict__ sumsq,
                       float* __restrict__ outscal) {
    __shared__ float red[256];
    int t = threadIdx.x;
    float e = 0.f;
    for (int k = t; k < Kq; k += 256) {
        float p = counts[k] * (1.0f / (float)Nq);
        e += p * logf(p + 1e-10f);
    }
    red[t] = e;
    __syncthreads();
    for (int s2 = 128; s2 > 0; s2 >>= 1) {
        if (t < s2) red[t] += red[t + s2];
        __syncthreads();
    }
    if (t == 0) {
        // codebook_loss == commitment_loss numerically -> (1 + BETA) * mse
        outscal[0] = 1.25f * (*sumsq) / (float)((size_t)Bq * Dq * Lq);
        outscal[1] = expf(-red[0]);
    }
}

extern "C" void kernel_launch(void* const* d_in, const int* in_sizes, int n_in,
                              void* d_out, int out_size, void* d_ws, size_t ws_size,
                              hipStream_t stream) {
    const float* z_e = (const float*)d_in[0];   // [32, 256, 1024] fp32
    const float* cb  = (const float*)d_in[1];   // [1024, 256] fp32

    // workspace layout
    _Float16* cb16  = (_Float16*)d_ws;                           // 512 KB
    float*    hnorm = (float*)((char*)d_ws + 524288);            // 4 KB
    float*    cnts  = (float*)((char*)d_ws + 528384);            // 4 KB
    float*    ssq   = (float*)((char*)d_ws + 532480);            // 4 B

    float* out  = (float*)d_out;
    float* scal = out + (size_t)Bq * Dq * Lq;   // [vq_loss, perplexity]
    float* idxo = scal + 2;                     // indices as float

    vq_prep<<<(Kq + 255) / 256, 256, 0, stream>>>(cb, cb16, hnorm, cnts, ssq);
    vq_main<<<Nq / 128, 256, 0, stream>>>(z_e, cb, cb16, hnorm, cnts, ssq, out, idxo);
    vq_fin<<<1, 256, 0, stream>>>(cnts, ssq, scal);
}
